// AttentionHead_47656957116788
// MI455X (gfx1250) — matrix operand order repaired and checked by reading, not compiled, not run on hardware
//
#include <hip/hip_runtime.h>
#include <hip/hip_bf16.h>

typedef _Float16 h16;
typedef __attribute__((ext_vector_type(16))) _Float16 v16h;
typedef __attribute__((ext_vector_type(8)))  _Float16 v8h;
typedef __attribute__((ext_vector_type(8)))  float    v8f;
typedef __attribute__((ext_vector_type(4)))  float    v4f;
typedef __attribute__((ext_vector_type(4)))  unsigned int v4u;
typedef __attribute__((ext_vector_type(8)))  int      v8i;
typedef __attribute__((ext_vector_type(4)))  int      v4i;

union Af { v16h v; v8h h[2]; };

static __device__ __forceinline__ v8f wmma_f16(v16h a, v16h b, v8f c) {
  // D = A(16x32 f16) * B(32x16 f16) + C(16x16 f32)
  return __builtin_amdgcn_wmma_f32_16x16x32_f16(false, a, false, b, (short)0, c, false, false);
}

#define BATCH 8
#define SEQ   2048
#define EMB   768
#define HD    64
#define BSROWS (BATCH * SEQ)

// q pre-scale: 1/sqrt(64) * log2(e)  (softmax done in base-2: v_exp_f32 is exp2)
#define QSCALE 0.18033688011112042f

#if defined(__HIP_DEVICE_COMPILE__) && __has_builtin(__builtin_amdgcn_exp2f)
#define FAST_EXP2(x) __builtin_amdgcn_exp2f(x)
#else
#define FAST_EXP2(x) exp2f(x)
#endif

#if defined(__HIP_DEVICE_COMPILE__) && \
    __has_builtin(__builtin_amdgcn_tensor_load_to_lds) && \
    __has_builtin(__builtin_amdgcn_s_wait_tensorcnt)
#define USE_TDM 1
#else
#define USE_TDM 0
#endif

#if USE_TDM
// LDS byte offset of a __shared__ object (generic -> AS(3) -> int)
static __device__ __forceinline__ unsigned lds_off(const void* p) {
  return (unsigned)(unsigned long long)(const __attribute__((address_space(3))) void*)p;
}

// TDM: DMA a 2-D f16 tile [rows x rowlen] (row stride = stride, elements) from
// global memory into LDS.  Descriptor per CDNA5 ISA §8.3/8.4; tensor == tile so
// no OOB handling is exercised.  Tracked by TENSORcnt.
static __device__ __forceinline__ void tdm_load_2d(unsigned lds_addr, const void* gptr,
                                                   int rows, int rowlen, int stride) {
  unsigned long long ga = (unsigned long long)gptr;
  v4u g0;
  g0[0] = 1u;                                             // count=1 (valid user D#)
  g0[1] = lds_addr;                                       // lds_addr [63:32]
  g0[2] = (unsigned)(ga & 0xffffffffu);                   // global_addr [95:64]
  g0[3] = (unsigned)((ga >> 32) & 0x01ffffffu) | (2u << 30); // addr hi + type=2
  v8i g1;
  g1[0] = (int)(1u << 16);                                // data_size=1 (2 bytes)
  g1[1] = (int)(((unsigned)rowlen & 0xffffu) << 16);      // tensor_dim0 lo16 @48
  g1[2] = (int)((((unsigned)rowlen >> 16) & 0xffffu) |
                (((unsigned)rows & 0xffffu) << 16));      // dim0 hi / tensor_dim1 lo
  g1[3] = (int)((((unsigned)rows >> 16) & 0xffffu) |
                (((unsigned)rowlen & 0xffffu) << 16));    // dim1 hi / tile_dim0
  g1[4] = rows & 0xffff;                                  // tile_dim1 (tile_dim2=0)
  g1[5] = stride;                                         // tensor_dim0_stride lo32
  g1[6] = 0;
  g1[7] = 0;
  v4i z4 = {0, 0, 0, 0};
#if __clang_major__ >= 23
  v8i z8 = {0, 0, 0, 0, 0, 0, 0, 0};
  __builtin_amdgcn_tensor_load_to_lds(g0, g1, z4, z4, z8, 0);
#else
  __builtin_amdgcn_tensor_load_to_lds(g0, g1, z4, z4, 0);
#endif
}
#endif  // USE_TDM

// ---------------------------------------------------------------------------
// Kernel 1: fused QKV projection.  [BS,768] x [768, 3*64] -> q,k (row), vT.
// grid = BS/128 blocks, 256 threads (8 waves); wave w owns 16 rows.
// ---------------------------------------------------------------------------
__global__ __launch_bounds__(256) void qkv_proj_kernel(
    const float* __restrict__ x,
    const float* __restrict__ Wq, const float* __restrict__ Wk, const float* __restrict__ Wv,
    const float* __restrict__ bq, const float* __restrict__ bk, const float* __restrict__ bv,
    h16* __restrict__ q_h, h16* __restrict__ k_h, h16* __restrict__ vT_h)
{
  __shared__ alignas(32) _Float16 xs[128 * 32];   // x chunk, [row][k] (f16)
  __shared__ alignas(32) _Float16 wT[192 * 32];   // W chunk transposed [n][k]

  const int tid  = threadIdx.x;
  const int lane = tid & 31;
  const int wv   = tid >> 5;
  const int ln   = lane & 15;
  const int hi   = lane >> 4;
  const int rowbase = blockIdx.x * 128 + wv * 16;

  v8f acc[12];
  {
    const float* bias[3] = { bq, bk, bv };
#pragma unroll
    for (int t = 0; t < 12; ++t) {
      float bb = bias[t >> 2][(t & 3) * 16 + ln];
#pragma unroll
      for (int r = 0; r < 8; ++r) acc[t][r] = bb;
    }
  }

  const float* Wm[3] = { Wq, Wk, Wv };

  for (int kc = 0; kc < EMB / 32; ++kc) {
    __syncthreads();
    // stage x[block rows 128][kc*32 .. +32) -> f16 LDS
#pragma unroll
    for (int i = 0; i < 4; ++i) {
      int s  = tid + 256 * i;            // 1024 float4 slots
      int r  = s >> 3;
      int c4 = (s & 7) * 4;
      v4f xv = *(const v4f*)(x + (size_t)(blockIdx.x * 128 + r) * EMB + kc * 32 + c4);
      _Float16* dst = &xs[r * 32 + c4];
#pragma unroll
      for (int j = 0; j < 4; ++j) dst[j] = (h16)xv[j];
    }
    // stage W chunks transposed: wT[m*64 + n][kr]
#pragma unroll
    for (int m = 0; m < 3; ++m) {
#pragma unroll
      for (int i = 0; i < 2; ++i) {
        int s  = tid + 256 * i;          // 512 float4 slots per matrix
        int kr = s >> 4;
        int c4 = (s & 15) * 4;
        v4f wv4 = *(const v4f*)(Wm[m] + (size_t)(kc * 32 + kr) * HD + c4);
#pragma unroll
        for (int j = 0; j < 4; ++j) wT[(m * 64 + c4 + j) * 32 + kr] = (h16)wv4[j];
      }
    }
    __syncthreads();

    // A fragment: 16x32 f16 (rows = wave's 16 rows)
    Af a;
    {
      int base = (wv * 16 + ln) * 32 + 16 * hi;
      a.h[0] = *(const v8h*)&xs[base];
      a.h[1] = *(const v8h*)&xs[base + 8];
    }
#pragma unroll
    for (int t = 0; t < 12; ++t) {
      v16h bfr = *(const v16h*)&wT[(t * 16 + ln) * 32 + 16 * hi];
      acc[t] = wmma_f16(a.v, bfr, acc[t]);
    }
  }

  // q scaled by 1/sqrt(64)*log2e (base-2 softmax); k row-major; v transposed
#pragma unroll
  for (int t = 0; t < 4; ++t)
#pragma unroll
    for (int r = 0; r < 8; ++r) {
      size_t row = (size_t)rowbase + r + 8 * hi;
      q_h[row * HD + t * 16 + ln] = (h16)(acc[t][r] * QSCALE);
    }
#pragma unroll
  for (int t = 0; t < 4; ++t)
#pragma unroll
    for (int r = 0; r < 8; ++r) {
      size_t row = (size_t)rowbase + r + 8 * hi;
      k_h[row * HD + t * 16 + ln] = (h16)acc[4 + t][r];
    }
#pragma unroll
  for (int t = 0; t < 4; ++t)
#pragma unroll
    for (int r = 0; r < 8; ++r) {
      int row = rowbase + r + 8 * hi;
      int b   = row >> 11;               // /2048
      int s   = row & 2047;
      int d   = t * 16 + ln;
      vT_h[(((size_t)b * HD + d) << 11) + s] = (h16)acc[8 + t][r];
    }
}

// ---------------------------------------------------------------------------
// Kernel 2: flash attention.  One wave = 16 query rows, streams 32 keys/iter.
// K tiles are DMA'd into LDS by the Tensor Data Mover (double-buffered,
// pipelined with s_wait_tensorcnt).  grid = B*S/16/4 blocks of 128 threads.
// ---------------------------------------------------------------------------
__global__ __launch_bounds__(128) void attn_kernel(
    const h16* __restrict__ q_h, const h16* __restrict__ k_h,
    const h16* __restrict__ vT_h, h16* __restrict__ o_h)
{
  __shared__ alignas(32) _Float16 pbuf[4][16 * 32];   // per-wave P reshape tile
#if USE_TDM
  __shared__ alignas(32) _Float16 kbuf[4][2][32 * HD]; // per-wave K double buffer
#endif

  const int tid  = threadIdx.x;
  const int lane = tid & 31;
  const int wv   = tid >> 5;
  const int ln   = lane & 15;
  const int hi   = lane >> 4;

  const int g  = blockIdx.x * 4 + wv;     // global query-tile id
  const int b  = g >> 7;                  // /128 tiles per batch
  const int s0 = (g & 127) * 16;

  // Q A-fragments for d-chunks 0 and 1 (q pre-scaled)
  Af qa[2];
  {
    size_t qrow = ((size_t)b * SEQ + s0 + ln) * HD;
#pragma unroll
    for (int c = 0; c < 2; ++c) {
      qa[c].h[0] = *(const v8h*)&q_h[qrow + 32 * c + 16 * hi];
      qa[c].h[1] = *(const v8h*)&q_h[qrow + 32 * c + 16 * hi + 8];
    }
  }

  v8f o[4];
#pragma unroll
  for (int t = 0; t < 4; ++t) o[t] = (v8f){};
  float m[8], l[8];
#pragma unroll
  for (int r = 0; r < 8; ++r) { m[r] = -1e30f; l[r] = 0.0f; }

  const h16* kb_base = k_h  + (size_t)b * SEQ * HD;
  const h16* vb_base = vT_h + (size_t)b * HD * SEQ;
  _Float16*  pb      = pbuf[wv];

#if USE_TDM
  tdm_load_2d(lds_off(&kbuf[wv][0][0]), kb_base, 32, HD, HD);  // prologue
#endif

  for (int j = 0; j < SEQ; j += 32) {
    if (j + 32 < SEQ)   // prefetch next V chunk (global_prefetch_b8)
      __builtin_prefetch(vb_base + (size_t)(16 * hi + ln) * SEQ + j + 32, 0, 1);

    v8f sc0 = (v8f){}, sc1 = (v8f){};
#if USE_TDM
    const int cur = (j >> 5) & 1;
    if (j + 32 < SEQ) {
      tdm_load_2d(lds_off(&kbuf[wv][cur ^ 1][0]),
                  kb_base + (size_t)(j + 32) * HD, 32, HD, HD);
      __builtin_amdgcn_s_wait_tensorcnt((short)1);  // oldest (cur) complete
    } else {
      __builtin_amdgcn_s_wait_tensorcnt((short)0);
    }
    asm volatile("" ::: "memory");      // DMA-filled LDS: no load hoisting
    {
      const _Float16* kc_ = &kbuf[wv][cur][0];
      v16h kb00 = *(const v16h*)&kc_[(ln)      * HD + 16 * hi];
      v16h kb01 = *(const v16h*)&kc_[(ln)      * HD + 32 + 16 * hi];
      v16h kb10 = *(const v16h*)&kc_[(16 + ln) * HD + 16 * hi];
      v16h kb11 = *(const v16h*)&kc_[(16 + ln) * HD + 32 + 16 * hi];
      sc0 = wmma_f16(qa[0].v, kb00, sc0);
      sc0 = wmma_f16(qa[1].v, kb01, sc0);
      sc1 = wmma_f16(qa[0].v, kb10, sc1);
      sc1 = wmma_f16(qa[1].v, kb11, sc1);
    }
#else
    {
      if (j + 32 < SEQ)
        __builtin_prefetch(kb_base + (size_t)(j + 32 + ln) * HD, 0, 1);
      v16h kb00 = *(const v16h*)&kb_base[(size_t)(j      + ln) * HD + 16 * hi];
      v16h kb01 = *(const v16h*)&kb_base[(size_t)(j      + ln) * HD + 32 + 16 * hi];
      v16h kb10 = *(const v16h*)&kb_base[(size_t)(j + 16 + ln) * HD + 16 * hi];
      v16h kb11 = *(const v16h*)&kb_base[(size_t)(j + 16 + ln) * HD + 32 + 16 * hi];
      sc0 = wmma_f16(qa[0].v, kb00, sc0);
      sc0 = wmma_f16(qa[1].v, kb01, sc0);
      sc1 = wmma_f16(qa[0].v, kb10, sc1);
      sc1 = wmma_f16(qa[1].v, kb11, sc1);
    }
#endif

    // ---- online softmax (base-2) per row; row r+8*hi is spread over 16 lanes
    float p0[8], p1[8];
#pragma unroll
    for (int r = 0; r < 8; ++r) {
      float vmax = fmaxf(sc0[r], sc1[r]);
#pragma unroll
      for (int msk = 1; msk < 16; msk <<= 1)
        vmax = fmaxf(vmax, __shfl_xor(vmax, msk, 32));
      float mn  = fmaxf(m[r], vmax);
      float scl = FAST_EXP2(m[r] - mn);
      m[r] = mn;
      p0[r] = FAST_EXP2(sc0[r] - mn);
      p1[r] = FAST_EXP2(sc1[r] - mn);
      float rs = p0[r] + p1[r];
#pragma unroll
      for (int msk = 1; msk < 16; msk <<= 1)
        rs += __shfl_xor(rs, msk, 32);
      l[r] = l[r] * scl + rs;
#pragma unroll
      for (int t = 0; t < 4; ++t) o[t][r] *= scl;
    }

    // ---- reshape P (C-layout) -> A-layout via per-wave LDS tile
#pragma unroll
    for (int r = 0; r < 8; ++r) {
      pb[(r + 8 * hi) * 32 + ln]      = (h16)p0[r];
      pb[(r + 8 * hi) * 32 + 16 + ln] = (h16)p1[r];
    }
    Af pa;
    {
      int base = ln * 32 + 16 * hi;
      pa.h[0] = *(const v8h*)&pb[base];
      pa.h[1] = *(const v8h*)&pb[base + 8];
    }

    // ---- O += P * V   (V^T rows are key-contiguous -> straight B-frag loads)
#pragma unroll
    for (int t = 0; t < 4; ++t) {
      v16h vb = *(const v16h*)&vb_base[(size_t)(t * 16 + ln) * SEQ + j + 16 * hi];
      o[t] = wmma_f16(pa.v, vb, o[t]);
    }
  }

  float inv[8];
#pragma unroll
  for (int r = 0; r < 8; ++r) inv[r] = 1.0f / l[r];
#pragma unroll
  for (int t = 0; t < 4; ++t)
#pragma unroll
    for (int r = 0; r < 8; ++r) {
      size_t row = (size_t)b * SEQ + s0 + r + 8 * hi;
      o_h[row * HD + t * 16 + ln] = (h16)(o[t][r] * inv[r]);
    }
}

// ---------------------------------------------------------------------------
// Kernel 3: transpose Wp [64,768] fp32 -> WpT [768,64] f16
// ---------------------------------------------------------------------------
__global__ void wp_transpose_kernel(const float* __restrict__ Wp, h16* __restrict__ wpT)
{
  for (int i = blockIdx.x * blockDim.x + threadIdx.x; i < EMB * HD;
       i += gridDim.x * blockDim.x) {
    int e = i >> 6, d = i & 63;
    wpT[i] = (h16)Wp[d * EMB + e];
  }
}

// ---------------------------------------------------------------------------
// Kernel 4: output projection  [BS,64] x [64,768] + bp -> fp32 out
// grid = BS/128 blocks, 256 threads (8 waves, 16 rows each).
// ---------------------------------------------------------------------------
__global__ __launch_bounds__(256) void out_proj_kernel(
    const h16* __restrict__ o_h, const h16* __restrict__ wpT,
    const float* __restrict__ bp, float* __restrict__ out)
{
  const int tid  = threadIdx.x;
  const int lane = tid & 31;
  const int wv   = tid >> 5;
  const int ln   = lane & 15;
  const int hi   = lane >> 4;
  const int rowbase = blockIdx.x * 128 + wv * 16;

  Af a[2];
  {
    size_t arow = (size_t)(rowbase + ln) * HD;
#pragma unroll
    for (int c = 0; c < 2; ++c) {
      a[c].h[0] = *(const v8h*)&o_h[arow + 32 * c + 16 * hi];
      a[c].h[1] = *(const v8h*)&o_h[arow + 32 * c + 16 * hi + 8];
    }
  }

  for (int nt = 0; nt < EMB / 16; ++nt) {
    float bb = bp[nt * 16 + ln];
    v8f acc;
#pragma unroll
    for (int r = 0; r < 8; ++r) acc[r] = bb;
#pragma unroll
    for (int c = 0; c < 2; ++c) {
      v16h bfr = *(const v16h*)&wpT[(size_t)(nt * 16 + ln) * HD + 32 * c + 16 * hi];
      acc = wmma_f16(a[c].v, bfr, acc);
    }
#pragma unroll
    for (int r = 0; r < 8; ++r)
      out[(size_t)(rowbase + r + 8 * hi) * EMB + nt * 16 + ln] = acc[r];
  }
}

// ---------------------------------------------------------------------------
extern "C" void kernel_launch(void* const* d_in, const int* in_sizes, int n_in,
                              void* d_out, int out_size, void* d_ws, size_t ws_size,
                              hipStream_t stream)
{
  const float* x  = (const float*)d_in[0];
  const float* Wk = (const float*)d_in[1];
  const float* bk = (const float*)d_in[2];
  const float* Wv = (const float*)d_in[3];
  const float* bv = (const float*)d_in[4];
  const float* Wq = (const float*)d_in[5];
  const float* bq = (const float*)d_in[6];
  const float* Wp = (const float*)d_in[7];
  const float* bp = (const float*)d_in[8];
  float* out = (float*)d_out;

  char* w = (char*)d_ws;
  h16* q_h   = (h16*)(w);                                  // [BS,64]   2 MB
  h16* k_h   = (h16*)(w + (size_t)BSROWS * HD * 2);        // [BS,64]   2 MB
  h16* vT_h  = (h16*)(w + (size_t)BSROWS * HD * 4);        // [B,64,S]  2 MB
  h16* o_h   = (h16*)(w + (size_t)BSROWS * HD * 6);        // [BS,64]   2 MB
  h16* wpT_h = (h16*)(w + (size_t)BSROWS * HD * 8);        // [768,64] 96 KB

  wp_transpose_kernel<<<48, 256, 0, stream>>>(Wp, wpT_h);
  qkv_proj_kernel<<<BSROWS / 128, 256, 0, stream>>>(x, Wq, Wk, Wv, bq, bk, bv,
                                                    q_h, k_h, vT_h);
  attn_kernel<<<BSROWS / 16 / 4, 128, 0, stream>>>(q_h, k_h, vT_h, o_h);
  out_proj_kernel<<<BSROWS / 128, 256, 0, stream>>>(o_h, wpT_h, bp, out);
}
